// GroupQueryAttention_90838558311150
// MI455X (gfx1250) — compile-verified, matrix operand order checked by reading, not run on hardware
//
#include <hip/hip_runtime.h>

// ---------------------------------------------------------------------------
// GQA attention for MI455X (gfx1250, wave32). BF16 WMMA everywhere:
//   v_wmma_f32_16x16x32_bf16, fp32 accumulation. Software-pipelined fragment
//   loads so WMMAs overlap VMEM/DS instead of s_wait_loadcnt 0 per step.
// ---------------------------------------------------------------------------

typedef __attribute__((ext_vector_type(16))) __bf16 v16bf;
typedef __attribute__((ext_vector_type(8)))  float  v8f;

constexpr int B_  = 2;
constexpr int S_  = 2048;
constexpr int D_  = 2048;
constexpr int H_  = 16;
constexpr int KVH_ = 4;
constexpr int DH_ = 128;
constexpr int REP_ = H_ / KVH_;                // 4
constexpr float EPS_ = 1e-6f;
constexpr float QSCALE_ = 0.08838834764831845f; // 1/sqrt(128)
constexpr float NEG_LN_BASE_OVER_HALF_ = -0.14391156856f; // -ln(10000)/64

union Frag {
    v16bf v;
    uint4 u[2];
};

__device__ __forceinline__ v8f wmma_bf16(const Frag& a, const Frag& b, v8f c) {
    // D = A(16x32 bf16) * B(32x16 bf16) + C(16x16 f32)
    return __builtin_amdgcn_wmma_f32_16x16x32_bf16(
        /*neg_a=*/false, a.v, /*neg_b=*/false, b.v,
        /*c_mod=*/(short)0, c, /*reuse_a=*/false, /*reuse_b=*/false);
}

// ---------------------------------------------------------------------------
// fp32 -> bf16 elementwise conversion (grid-stride)
// ---------------------------------------------------------------------------
__global__ void gqa_cvt_f32_bf16(const float* __restrict__ in,
                                 __bf16* __restrict__ out, size_t n) {
    size_t i = (size_t)blockIdx.x * blockDim.x + threadIdx.x;
    size_t stride = (size_t)gridDim.x * blockDim.x;
    for (; i < n; i += stride) out[i] = (__bf16)in[i];
}

// ---------------------------------------------------------------------------
// C[M,N] (f32) = A[M,K] (bf16, row-major) * W[N,K]^T (bf16, row-major)
// Wave computes a 16x64 tile (4 accumulators); block of 8 waves covers 16x512.
// Double-buffered k-steps: loads for step i+1 issue before WMMAs of step i.
// grid = (M/16, N/512), block = 256. K must be a multiple of 64.
// ---------------------------------------------------------------------------
__global__ __launch_bounds__(256)
void gqa_gemm_bf16_nt(const __bf16* __restrict__ A, const __bf16* __restrict__ W,
                      float* __restrict__ C, int M, int N, int K) {
    const int lane = threadIdx.x & 31;
    const int wave = threadIdx.x >> 5;
    const int m0 = blockIdx.x * 16;
    const int n0 = blockIdx.y * 512 + wave * 64;
    const int r  = lane & 15;
    const int hf = lane >> 4;          // half-wave selects K sub-block per ISA layout

    v8f acc[4] = {};
    const __bf16* arow = A + (size_t)(m0 + r) * K;
    const __bf16* wrow[4];
#pragma unroll
    for (int t = 0; t < 4; ++t) wrow[t] = W + (size_t)(n0 + t * 16 + r) * K;

    Frag a[2], w[2][4];

    auto loadStep = [&](int buf, int k0) {
        __builtin_prefetch(arow + k0 + 256, 0, 3);
        a[buf].u[0] = *(const uint4*)(arow + k0 + hf * 8);
        a[buf].u[1] = *(const uint4*)(arow + k0 + 16 + hf * 8);
#pragma unroll
        for (int t = 0; t < 4; ++t) {
            __builtin_prefetch(wrow[t] + k0 + 256, 0, 3);
            w[buf][t].u[0] = *(const uint4*)(wrow[t] + k0 + hf * 8);
            w[buf][t].u[1] = *(const uint4*)(wrow[t] + k0 + 16 + hf * 8);
        }
    };
    auto compute = [&](int buf) {
#pragma unroll
        for (int t = 0; t < 4; ++t)
            acc[t] = wmma_bf16(a[buf], w[buf][t], acc[t]);
    };

    loadStep(0, 0);
    for (int k0 = 0; k0 + 64 <= K; k0 += 64) {
        loadStep(1, k0 + 32);          // overlap with compute(0)
        compute(0);
        if (k0 + 64 < K) loadStep(0, k0 + 64);  // overlap with compute(1)
        compute(1);
    }

    const int rbase = m0 + (hf ? 8 : 0);
#pragma unroll
    for (int t = 0; t < 4; ++t) {
        const int n = n0 + t * 16 + r;
#pragma unroll
        for (int j = 0; j < 8; ++j)
            C[(size_t)(rbase + j) * N + n] = acc[t][j];
    }
}

// ---------------------------------------------------------------------------
// Fused RMSNorm + RoPE over DH=128 per (b,s,head) row; one wave per row.
// Lane owns dims {l, l+32} paired with {l+64, l+96}. outScale folds 1/sqrt(DH)
// for Q. Writes bf16.
// ---------------------------------------------------------------------------
__global__ __launch_bounds__(256)
void gqa_normrope(const float* __restrict__ in, __bf16* __restrict__ out,
                  int rows, int heads, float outScale) {
    const int row = blockIdx.x * 8 + (threadIdx.x >> 5);
    if (row >= rows) return;
    const int lane = threadIdx.x & 31;
    const int s = (row / heads) % S_;

    const float* p = in + (size_t)row * DH_;
    float x0 = p[lane];        // dim l
    float x2 = p[lane + 32];   // dim l+32
    float x1 = p[lane + 64];   // dim l+64 (pair of x0)
    float x3 = p[lane + 96];   // dim l+96 (pair of x2)

    float ss = x0 * x0 + x1 * x1 + x2 * x2 + x3 * x3;
#pragma unroll
    for (int off = 1; off < 32; off <<= 1) ss += __shfl_xor(ss, off, 32);
    const float rn = rsqrtf(ss * (1.0f / DH_) + EPS_) * outScale;

    const float f0 = __expf(NEG_LN_BASE_OVER_HALF_ * (float)lane);
    const float f1 = __expf(NEG_LN_BASE_OVER_HALF_ * (float)(lane + 32));
    const float a0 = (float)s * f0, a1 = (float)s * f1;
    const float c0 = cosf(a0), s0 = sinf(a0);
    const float c1 = cosf(a1), s1 = sinf(a1);

    __bf16* q = out + (size_t)row * DH_;
    q[lane]      = (__bf16)((x0 * c0 - x1 * s0) * rn);
    q[lane + 64] = (__bf16)((x0 * s0 + x1 * c0) * rn);
    q[lane + 32] = (__bf16)((x2 * c1 - x3 * s1) * rn);
    q[lane + 96] = (__bf16)((x2 * s1 + x3 * c1) * rn);
}

// ---------------------------------------------------------------------------
// Causal flash attention. grid = (B*H, S/64), block = 128 (4 waves).
// Wave owns 16 query rows; block stages 32-key K tile + transposed V tile in
// LDS; online softmax with WMMA for both QK^T and PV. P transpose
// (accumulator layout -> A-fragment layout) goes through per-wave LDS.
// LDS-fed WMMA chains are double-buffered (distinct fragment registers).
// ---------------------------------------------------------------------------
__global__ __launch_bounds__(128)
void gqa_flash(const __bf16* __restrict__ Q, const __bf16* __restrict__ Kc,
               const __bf16* __restrict__ Vc, __bf16* __restrict__ O) {
    __shared__ __align__(16) __bf16 Kl[32 * 128];     // [key][dim]
    __shared__ __align__(16) __bf16 Vt[128 * 32];     // [dim][key]
    __shared__ __align__(16) __bf16 Pl[4][16 * 32];   // per-wave P tile

    const int bh = blockIdx.x;
    const int b = bh / H_, h = bh % H_;
    const int kvh = h / REP_;
    const int qtile = blockIdx.y;
    const int wave = threadIdx.x >> 5;
    const int lane = threadIdx.x & 31;
    const int r  = lane & 15;
    const int hf = lane >> 4;
    const int qbase = qtile * 64 + wave * 16;

    // Q fragments: 16 rows x 128 dims = 4 K-chunks of 32
    Frag qf[4];
    const __bf16* qrow = Q + ((size_t)(b * S_ + qbase + r) * H_ + h) * DH_;
#pragma unroll
    for (int c = 0; c < 4; ++c) {
        qf[c].u[0] = *(const uint4*)(qrow + c * 32 + hf * 8);
        qf[c].u[1] = *(const uint4*)(qrow + c * 32 + 16 + hf * 8);
    }

    v8f o[8] = {};
    float mrow[8], lrow[8], alpha[8];
#pragma unroll
    for (int j = 0; j < 8; ++j) { mrow[j] = -1e30f; lrow[j] = 0.0f; }

    const int tid = threadIdx.x;
    const int skey = tid >> 2;          // 0..31 : key row staged by this thread
    const int dseg = (tid & 3) * 32;    // 0,32,64,96 : dim segment
    const int nt = qtile * 2 + 2;       // key tiles needed by this block

    auto loadK = [&](Frag& f, int c, int half16) {
        const __bf16* kr = Kl + (size_t)(half16 * 16 + r) * 128 + c * 32;
        f.u[0] = *(const uint4*)(kr + hf * 8);
        f.u[1] = *(const uint4*)(kr + 16 + hf * 8);
    };
    auto loadV = [&](Frag& f, int fi) {
        const __bf16* vr = Vt + (size_t)(fi * 16 + r) * 32;
        f.u[0] = *(const uint4*)(vr + hf * 8);
        f.u[1] = *(const uint4*)(vr + 16 + hf * 8);
    };

    for (int t = 0; t < nt; ++t) {
        const int kb = t * 32;
        // ---- stage K tile and transposed V tile ----
        {
            const size_t kvoff = ((size_t)(b * S_ + kb + skey) * KVH_ + kvh) * DH_ + dseg;
            const uint4* ksrc = (const uint4*)(Kc + kvoff);
            uint4* kdst = (uint4*)(Kl + skey * 128 + dseg);
#pragma unroll
            for (int u = 0; u < 4; ++u) kdst[u] = ksrc[u];
            __bf16 vtmp[32];
            const uint4* vsrc = (const uint4*)(Vc + kvoff);
#pragma unroll
            for (int u = 0; u < 4; ++u) ((uint4*)vtmp)[u] = vsrc[u];
#pragma unroll
            for (int d = 0; d < 32; ++d) Vt[(dseg + d) * 32 + skey] = vtmp[d];
        }
        __syncthreads();

        if (kb <= qbase + 15) {   // wave-uniform
            // ---- scores: S = Q * K^T for 32 keys, double-buffered frags ----
            v8f slo = {}, shi = {};
            Frag bk[2];
            loadK(bk[0], 0, 0);
#pragma unroll
            for (int i = 0; i < 8; ++i) {
                if (i + 1 < 8) loadK(bk[(i + 1) & 1], (i + 1) >> 1, (i + 1) & 1);
                if (i & 1) shi = wmma_bf16(qf[i >> 1], bk[i & 1], shi);
                else       slo = wmma_bf16(qf[i >> 1], bk[i & 1], slo);
            }

            // ---- causal mask + online softmax (half-wave row reductions) ----
            const int rbase = qbase + (hf ? 8 : 0);
#pragma unroll
            for (int j = 0; j < 8; ++j) {
                const int qg = rbase + j;
                float sl = (kb + r      <= qg) ? slo[j] : -1e30f;
                float sh = (kb + 16 + r <= qg) ? shi[j] : -1e30f;
                float mx = fmaxf(sl, sh);
#pragma unroll
                for (int off = 1; off < 16; off <<= 1)
                    mx = fmaxf(mx, __shfl_xor(mx, off, 32));
                const float mnew = fmaxf(mrow[j], mx);
                const float pl = __expf(sl - mnew);
                const float ph = __expf(sh - mnew);
                float rs = pl + ph;
#pragma unroll
                for (int off = 1; off < 16; off <<= 1)
                    rs += __shfl_xor(rs, off, 32);
                alpha[j] = __expf(mrow[j] - mnew);
                lrow[j] = lrow[j] * alpha[j] + rs;
                mrow[j] = mnew;
                // spill P to LDS (accumulator layout -> A-fragment layout)
                const int prow = j + (hf ? 8 : 0);
                Pl[wave][prow * 32 + r]      = (__bf16)pl;
                Pl[wave][prow * 32 + 16 + r] = (__bf16)ph;
            }

            // ---- P fragment from LDS; O += P * V (double-buffered frags) ----
            Frag pf;
            const __bf16* pr = &Pl[wave][(size_t)r * 32];
            pf.u[0] = *(const uint4*)(pr + hf * 8);
            pf.u[1] = *(const uint4*)(pr + 16 + hf * 8);
            Frag bv[2];
            loadV(bv[0], 0);
#pragma unroll
            for (int f = 0; f < 8; ++f) {
                if (f + 1 < 8) loadV(bv[(f + 1) & 1], f + 1);
#pragma unroll
                for (int j = 0; j < 8; ++j) o[f][j] *= alpha[j];
                o[f] = wmma_bf16(pf, bv[f & 1], o[f]);
            }
        }
        __syncthreads();
    }

    // ---- epilogue: divide by row sums, store bf16 ----
    const int rbase = qbase + (hf ? 8 : 0);
#pragma unroll
    for (int j = 0; j < 8; ++j) {
        const float inv = 1.0f / lrow[j];
        const int qg = rbase + j;
        __bf16* orow = O + ((size_t)(b * S_ + qg) * H_ + h) * DH_;
#pragma unroll
        for (int f = 0; f < 8; ++f)
            orow[f * 16 + r] = (__bf16)(o[f][j] * inv);
    }
}

// ---------------------------------------------------------------------------
// Host-side orchestration
// ---------------------------------------------------------------------------
extern "C" void kernel_launch(void* const* d_in, const int* in_sizes, int n_in,
                              void* d_out, int out_size, void* d_ws, size_t ws_size,
                              hipStream_t stream) {
    (void)in_sizes; (void)n_in; (void)out_size; (void)ws_size;
    const float* x  = (const float*)d_in[0];
    // d_in[1] = mask: causal by construction, handled analytically
    const float* wq = (const float*)d_in[2];
    const float* wk = (const float*)d_in[3];
    const float* wv = (const float*)d_in[4];
    const float* wo = (const float*)d_in[5];
    float* out = (float*)d_out;

    const size_t MR = (size_t)B_ * S_;       // 4096 rows
    const size_t KVD = (size_t)KVH_ * DH_;   // 512

    char* ws = (char*)d_ws;
    size_t off = 0;
    auto alloc = [&](size_t bytes) -> char* {
        char* p = ws + off;
        off += (bytes + 255) & ~(size_t)255;
        return p;
    };
    __bf16* xb  = (__bf16*)alloc(MR * D_ * 2);
    __bf16* wqb = (__bf16*)alloc((size_t)D_ * D_ * 2);
    __bf16* wkb = (__bf16*)alloc(KVD * D_ * 2);
    __bf16* wvb = (__bf16*)alloc(KVD * D_ * 2);
    __bf16* wob = (__bf16*)alloc((size_t)D_ * D_ * 2);
    float*  qfp = (float*)alloc(MR * D_ * 4);
    float*  kfp = (float*)alloc(MR * KVD * 4);
    float*  vfp = (float*)alloc(MR * KVD * 4);
    __bf16* qb  = (__bf16*)alloc(MR * D_ * 2);
    __bf16* kb  = (__bf16*)alloc(MR * KVD * 2);
    __bf16* vb  = (__bf16*)alloc(MR * KVD * 2);
    __bf16* ab  = (__bf16*)alloc(MR * D_ * 2);

    // 1) fp32 -> bf16 (everything then lives in the 192MB L2)
    gqa_cvt_f32_bf16<<<2048, 256, 0, stream>>>(x,  xb,  MR * D_);
    gqa_cvt_f32_bf16<<<2048, 256, 0, stream>>>(wq, wqb, (size_t)D_ * D_);
    gqa_cvt_f32_bf16<<<1024, 256, 0, stream>>>(wk, wkb, KVD * D_);
    gqa_cvt_f32_bf16<<<1024, 256, 0, stream>>>(wv, wvb, KVD * D_);
    gqa_cvt_f32_bf16<<<2048, 256, 0, stream>>>(wo, wob, (size_t)D_ * D_);

    // 2) QKV projections (WMMA bf16, f32 accum)
    gqa_gemm_bf16_nt<<<dim3(MR / 16, D_ / 512), 256, 0, stream>>>(xb, wqb, qfp, MR, D_, D_);
    gqa_gemm_bf16_nt<<<dim3(MR / 16, 1),        256, 0, stream>>>(xb, wkb, kfp, MR, 512, D_);
    gqa_gemm_bf16_nt<<<dim3(MR / 16, 1),        256, 0, stream>>>(xb, wvb, vfp, MR, 512, D_);

    // 3) RMSNorm + RoPE (fold 1/sqrt(DH) into Q); V just converts to bf16
    gqa_normrope<<<(MR * H_) / 8,   256, 0, stream>>>(qfp, qb, MR * H_,   H_,   QSCALE_);
    gqa_normrope<<<(MR * KVH_) / 8, 256, 0, stream>>>(kfp, kb, MR * KVH_, KVH_, 1.0f);
    gqa_cvt_f32_bf16<<<1024, 256, 0, stream>>>(vfp, vb, MR * KVD);

    // 4) causal flash attention (WMMA for QK^T and PV)
    gqa_flash<<<dim3(B_ * H_, S_ / 64), 128, 0, stream>>>(qb, kb, vb, ab);

    // 5) output projection -> fp32 result
    gqa_gemm_bf16_nt<<<dim3(MR / 16, D_ / 512), 256, 0, stream>>>(ab, wob, out, MR, D_, D_);
}
